// BitLinear_12910671692515
// MI455X (gfx1250) — 8-bit matrix operand order repaired and checked by reading, compiled, not run on hardware
//
#include <hip/hip_runtime.h>

typedef int   v8i __attribute__((ext_vector_type(8)));
typedef int   v4i __attribute__((ext_vector_type(4)));
typedef int   v2i __attribute__((ext_vector_type(2)));
typedef float v4f __attribute__((ext_vector_type(4)));

#define M_TOK 8192
#define NF    4096
#define KF    4096
#define BK    64
#define LDSS  80
#define LDSTILE (128 * LDSS)
#define RED_BLOCKS 2048

#define AS1 __attribute__((address_space(1)))
#define AS3 __attribute__((address_space(3)))

#if defined(__has_builtin)
#if __has_builtin(__builtin_amdgcn_global_load_async_to_lds_b128)
#define USE_ASYNC_LDS 1
#endif
#if __has_builtin(__builtin_amdgcn_s_wait_asynccnt)
#define HAVE_WAIT_ASYNC_BUILTIN 1
#endif
#endif

__device__ __forceinline__ void wait_async0() {
#if defined(HAVE_WAIT_ASYNC_BUILTIN)
    __builtin_amdgcn_s_wait_asynccnt(0);
#else
    asm volatile("s_wait_asynccnt 0" ::: "memory");
#endif
}


__global__ void init_scalars(float* sumw, unsigned* maxbits) {
    if (threadIdx.x == 0) { *sumw = 0.0f; *maxbits = 0u; }
}

__global__ __launch_bounds__(256)
void absmax_kernel(const float* __restrict__ x, unsigned* maxbits, int n4) {
    __shared__ unsigned red[256];
    float m = 0.0f;
    int stride = gridDim.x * blockDim.x;
    for (int i = blockIdx.x * blockDim.x + threadIdx.x; i < n4; i += stride) {
        v4f v = ((const v4f*)x)[i];
        m = fmaxf(m, fmaxf(fmaxf(fabsf(v[0]), fabsf(v[1])),
                           fmaxf(fabsf(v[2]), fabsf(v[3]))));
    }
    red[threadIdx.x] = __float_as_uint(m);
    __syncthreads();
    for (int s = 128; s > 0; s >>= 1) {
        if (threadIdx.x < s)
            red[threadIdx.x] = red[threadIdx.x] > red[threadIdx.x + s] ?
                               red[threadIdx.x] : red[threadIdx.x + s];
        __syncthreads();
    }
    if (threadIdx.x == 0) atomicMax(maxbits, red[0]);
}

__global__ __launch_bounds__(256)
void sum_partial_kernel(const float* __restrict__ w, float* partials, int n4) {
    __shared__ float red[256];
    float s = 0.0f;
    int stride = gridDim.x * blockDim.x;
    for (int i = blockIdx.x * blockDim.x + threadIdx.x; i < n4; i += stride) {
        v4f v = ((const v4f*)w)[i];
        s += (v[0] + v[1]) + (v[2] + v[3]);
    }
    red[threadIdx.x] = s;
    __syncthreads();
    for (int t = 128; t > 0; t >>= 1) {
        if (threadIdx.x < t) red[threadIdx.x] += red[threadIdx.x + t];
        __syncthreads();
    }
    if (threadIdx.x == 0) partials[blockIdx.x] = red[0];
}

__global__ __launch_bounds__(256)
void sum_final_kernel(const float* __restrict__ partials, float* sumw, int n) {
    __shared__ float red[256];
    float s = 0.0f;
    for (int i = threadIdx.x; i < n; i += 256) s += partials[i];
    red[threadIdx.x] = s;
    __syncthreads();
    for (int t = 128; t > 0; t >>= 1) {
        if (threadIdx.x < t) red[threadIdx.x] += red[threadIdx.x + t];
        __syncthreads();
    }
    if (threadIdx.x == 0) *sumw = red[0];
}

__global__ __launch_bounds__(256)
void quantize_kernel(const float* __restrict__ x, signed char* __restrict__ q8,
                     const unsigned* __restrict__ maxbits, int n4) {
    float scale = 127.0f / __uint_as_float(*maxbits);
    int stride = gridDim.x * blockDim.x;
    for (int i = blockIdx.x * blockDim.x + threadIdx.x; i < n4; i += stride) {
        v4f v = ((const v4f*)x)[i];
        int b0 = __float2int_rn(scale * v[0]) & 0xff;
        int b1 = __float2int_rn(scale * v[1]) & 0xff;
        int b2 = __float2int_rn(scale * v[2]) & 0xff;
        int b3 = __float2int_rn(scale * v[3]) & 0xff;
        ((int*)q8)[i] = b0 | (b1 << 8) | (b2 << 16) | (b3 << 24);
    }
}

__global__ __launch_bounds__(256)
void ternarize_kernel(const float* __restrict__ w, signed char* __restrict__ s8,
                      const float* __restrict__ sumw, int n4) {
    float mean = (*sumw) * (1.0f / ((float)NF * (float)KF));
    int stride = gridDim.x * blockDim.x;
    for (int i = blockIdx.x * blockDim.x + threadIdx.x; i < n4; i += stride) {
        v4f v = ((const v4f*)w)[i];
        int b0 = ((v[0] > mean) - (v[0] < mean)) & 0xff;
        int b1 = ((v[1] > mean) - (v[1] < mean)) & 0xff;
        int b2 = ((v[2] > mean) - (v[2] < mean)) & 0xff;
        int b3 = ((v[3] > mean) - (v[3] < mean)) & 0xff;
        ((int*)s8)[i] = b0 | (b1 << 8) | (b2 << 16) | (b3 << 24);
    }
}

__global__ __launch_bounds__(256)
void bitlinear_gemm_kernel(const signed char* __restrict__ Q,
                           const signed char* __restrict__ S,
                           float* __restrict__ out) {
    __shared__ unsigned char ldsA[2 * LDSTILE];
    __shared__ unsigned char ldsB[2 * LDSTILE];
    const int tid    = threadIdx.x;
    const int lane   = tid & 31;
    const int wid    = tid >> 5;
    const int wave_m = wid >> 2;
    const int wave_n = wid & 3;
    const int lmod   = lane & 15;
    const int lhalf  = lane >> 4;
    const size_t m0 = (size_t)blockIdx.y * 128;
    const size_t n0 = (size_t)blockIdx.x * 128;

    const int row0 = tid >> 2;
    const int row1 = (tid + 256) >> 2;
    const int off0 = (tid & 3) * 16;
    const int off1 = ((tid + 256) & 3) * 16;

    v8i acc[4][2];
    #pragma unroll
    for (int i = 0; i < 4; i++)
        #pragma unroll
        for (int j = 0; j < 2; j++)
            acc[i][j] = (v8i){0, 0, 0, 0, 0, 0, 0, 0};

#if defined(USE_ASYNC_LDS)
    auto stage = [&](int it, int buf) {
        const size_t k0 = (size_t)it * BK;
        const unsigned base = (unsigned)buf * LDSTILE;
        __builtin_amdgcn_global_load_async_to_lds_b128(
            (AS1 v4i*)(v4i*)(Q + (m0 + row0) * KF + k0 + off0),
            (AS3 v4i*)(v4i*)&ldsA[base + row0 * LDSS + off0], 0, 0);
        __builtin_amdgcn_global_load_async_to_lds_b128(
            (AS1 v4i*)(v4i*)(Q + (m0 + row1) * KF + k0 + off1),
            (AS3 v4i*)(v4i*)&ldsA[base + row1 * LDSS + off1], 0, 0);
        __builtin_amdgcn_global_load_async_to_lds_b128(
            (AS1 v4i*)(v4i*)(S + (n0 + row0) * KF + k0 + off0),
            (AS3 v4i*)(v4i*)&ldsB[base + row0 * LDSS + off0], 0, 0);
        __builtin_amdgcn_global_load_async_to_lds_b128(
            (AS1 v4i*)(v4i*)(S + (n0 + row1) * KF + k0 + off1),
            (AS3 v4i*)(v4i*)&ldsB[base + row1 * LDSS + off1], 0, 0);
    };
#else
    v4i ra0, ra1, rb0, rb1;
    auto gload = [&](int it) {
        const size_t k0 = (size_t)it * BK;
        ra0 = *(const v4i*)(Q + (m0 + row0) * KF + k0 + off0);
        ra1 = *(const v4i*)(Q + (m0 + row1) * KF + k0 + off1);
        rb0 = *(const v4i*)(S + (n0 + row0) * KF + k0 + off0);
        rb1 = *(const v4i*)(S + (n0 + row1) * KF + k0 + off1);
    };
    auto lstore = [&](int buf) {
        const unsigned base = (unsigned)buf * LDSTILE;
        *(v4i*)&ldsA[base + row0 * LDSS + off0] = ra0;
        *(v4i*)&ldsA[base + row1 * LDSS + off1] = ra1;
        *(v4i*)&ldsB[base + row0 * LDSS + off0] = rb0;
        *(v4i*)&ldsB[base + row1 * LDSS + off1] = rb1;
    };
#endif

    auto compute = [&](int buf) {
        const unsigned base = (unsigned)buf * LDSTILE;
        v8i afrag[4];
        #pragma unroll
        for (int mt = 0; mt < 4; mt++) {
            int row = wave_m * 64 + mt * 16 + lmod;
            int kh  = lhalf * 16;
            #pragma unroll
            for (int g = 0; g < 4; g++) {
                v2i t = *(const v2i*)&ldsA[base + row * LDSS + kh + 8 * g + 16 * (g >> 1)];
                afrag[mt][2 * g]     = t[0];
                afrag[mt][2 * g + 1] = t[1];
            }
        }
        v8i bfrag[2];
        #pragma unroll
        for (int nt = 0; nt < 2; nt++) {
            int nrow = wave_n * 32 + nt * 16 + lmod;
            int ko   = lhalf * 16;
            v4i lo = *(const v4i*)&ldsB[base + nrow * LDSS + ko];
            v4i hi = *(const v4i*)&ldsB[base + nrow * LDSS + 32 + ko];
            #pragma unroll
            for (int j = 0; j < 4; j++) {
                bfrag[nt][j]     = lo[j];
                bfrag[nt][4 + j] = hi[j];
            }
        }
        #pragma unroll
        for (int mt = 0; mt < 4; mt++)
            #pragma unroll
            for (int nt = 0; nt < 2; nt++)
                acc[mt][nt] = __builtin_amdgcn_wmma_i32_16x16x64_iu8(
                    true, afrag[mt], true, bfrag[nt], acc[mt][nt], false, false);
    };

    const int NIT = KF / BK;

#if defined(USE_ASYNC_LDS)
    stage(0, 0);
    wait_async0();
#else
    gload(0);
    lstore(0);
#endif
    __syncthreads();

    for (int it = 0; it < NIT - 1; it++) {
        const int cur = it & 1;
        const int nxt = cur ^ 1;
        const int itp = (it + 2 < NIT) ? (it + 2) : (NIT - 1);
        __builtin_prefetch(Q + (m0 + row0) * KF + (size_t)itp * BK + off0, 0, 1);
        __builtin_prefetch(S + (n0 + row0) * KF + (size_t)itp * BK + off0, 0, 1);
#if defined(USE_ASYNC_LDS)
        stage(it + 1, nxt);
        compute(cur);
        wait_async0();
#else
        gload(it + 1);
        compute(cur);
        lstore(nxt);
#endif
        __syncthreads();
    }
    compute((NIT - 1) & 1);

    #pragma unroll
    for (int mt = 0; mt < 4; mt++) {
        #pragma unroll
        for (int nt = 0; nt < 2; nt++) {
            size_t gm0 = m0 + wave_m * 64 + mt * 16 + lhalf * 8;
            size_t gn  = n0 + wave_n * 32 + nt * 16 + lmod;
            #pragma unroll
            for (int r = 0; r < 8; r++)
                out[(gm0 + r) * NF + gn] = (float)acc[mt][nt][r];
        }
    }
}

extern "C" void kernel_launch(void* const* d_in, const int* in_sizes, int n_in,
                              void* d_out, int out_size, void* d_ws, size_t ws_size,
                              hipStream_t stream) {
    const float* x = (const float*)d_in[0];
    const float* w = (const float*)d_in[1];
    float* out = (float*)d_out;

    char* ws = (char*)d_ws;
    float*    sumw     = (float*)ws;
    unsigned* maxbits  = (unsigned*)(ws + 4);
    float*    partials = (float*)(ws + 64);
    signed char* s8 = (signed char*)(ws + 16384);
    signed char* q8 = (signed char*)(ws + 16384 + (size_t)NF * KF);

    const int nx4 = M_TOK * KF / 4;
    const int nw4 = NF * KF / 4;

    init_scalars<<<1, 64, 0, stream>>>(sumw, maxbits);
    absmax_kernel<<<RED_BLOCKS, 256, 0, stream>>>(x, maxbits, nx4);
    sum_partial_kernel<<<RED_BLOCKS, 256, 0, stream>>>(w, partials, nw4);
    sum_final_kernel<<<1, 256, 0, stream>>>(partials, sumw, RED_BLOCKS);
    quantize_kernel<<<RED_BLOCKS, 256, 0, stream>>>(x, q8, maxbits, nx4);
    ternarize_kernel<<<RED_BLOCKS, 256, 0, stream>>>(w, s8, sumw, nw4);

    dim3 grid(NF / 128, M_TOK / 128);
    bitlinear_gemm_kernel<<<grid, 256, 0, stream>>>(q8, s8, out);
}
